// LlamaAttention_49615462203846
// MI455X (gfx1250) — matrix operand order repaired and checked by reading, not compiled, not run on hardware
//
#include <hip/hip_runtime.h>

// ---------------------------------------------------------------------------
// Llama attention layer for MI455X (gfx1250, wave32, WMMA).
// All GEMM work runs on v_wmma_f32_16x16x32_bf16 (fp32 accumulate).
// Data movement uses the CDNA5 async-copy engine (global_load_async_to_lds,
// ASYNCcnt) with LDS double-buffering for the GEMMs and LDS-staged V tiles
// in the flash-attention kernel. GEMM wave tiles are 64x64 (16 WMMA/K-step)
// to keep the WMMA pipe ahead of LDS read bandwidth.
// Problem: B=2 S=2048 DIM=4096 NH=32 NKV=8 HD=128, paged KV (BLOCK=16).
// ---------------------------------------------------------------------------

typedef __attribute__((ext_vector_type(16))) __bf16 v16bf;
typedef __attribute__((ext_vector_type(8)))  float  v8f;

#define B_     2
#define S_     2048
#define DIM_   4096
#define NH_    32
#define NKV_   8
#define HD_    128
#define QKVC_  6144          // (NH + 2*NKV) * HD
#define NBLK_  128           // S / BLOCK
#define SCALE_ 0.08838834764831845f   // 1/sqrt(128)

// float -> bf16, round-to-nearest-even
__device__ __forceinline__ __bf16 f2bf(float f) {
    unsigned u = __builtin_bit_cast(unsigned, f);
    u += 0x7fffu + ((u >> 16) & 1u);
    unsigned short h = (unsigned short)(u >> 16);
    return __builtin_bit_cast(__bf16, h);
}

// CDNA5 async copy: each active lane copies 16B global -> LDS (ASYNCcnt).
__device__ __forceinline__ void async_copy_b128(void* lds_ptr, const void* gptr) {
    unsigned lds_off = (unsigned)(uintptr_t)lds_ptr;       // addr[31:0] = LDS offset
    unsigned long long ga = (unsigned long long)(uintptr_t)gptr;
    asm volatile("global_load_async_to_lds_b128 %0, %1, off"
                 :: "v"(lds_off), "v"(ga) : "memory");
}
__device__ __forceinline__ void wait_async0() {
    asm volatile("s_wait_asynccnt 0x0" ::: "memory");
}

// Row reduction across the 16 lanes that share a C-fragment row (wave32).
__device__ __forceinline__ float rmax16(float v) {
    v = fmaxf(v, __shfl_xor(v, 1, 32));
    v = fmaxf(v, __shfl_xor(v, 2, 32));
    v = fmaxf(v, __shfl_xor(v, 4, 32));
    v = fmaxf(v, __shfl_xor(v, 8, 32));
    return v;
}
__device__ __forceinline__ float rsum16(float v) {
    v += __shfl_xor(v, 1, 32);
    v += __shfl_xor(v, 2, 32);
    v += __shfl_xor(v, 4, 32);
    v += __shfl_xor(v, 8, 32);
    return v;
}

// A-fragment (16x32 bf16) from row-major [.,ld]: row = lane&15,
// K(e) = (e&7) + (e>>3)*16 + (lane>>4)*8
__device__ __forceinline__ v16bf load_a_frag(const __bf16* __restrict__ p,
                                             int ld, int row0, int k0, int lane) {
    const __bf16* base = p + (size_t)(row0 + (lane & 15)) * ld + k0 + ((lane >> 4) << 3);
    v16bf a;
#pragma unroll
    for (int e = 0; e < 8; ++e) a[e] = base[e];
#pragma unroll
    for (int e = 0; e < 8; ++e) a[8 + e] = base[16 + e];
    return a;
}

// ---------------------------------------------------------------------------
__global__ void __launch_bounds__(256)
cast_f32_bf16_kernel(const float* __restrict__ in, __bf16* __restrict__ out, size_t n) {
    size_t i = (size_t)blockIdx.x * blockDim.x + threadIdx.x;
    size_t stride = (size_t)gridDim.x * blockDim.x;
    for (; i < n; i += stride) out[i] = f2bf(in[i]);
}

// ---------------------------------------------------------------------------
// C[M,N] (f32) = A[M,K](bf16, row-major) @ W[N,K](bf16, row-major)^T
// 256 threads = 8 waves (2x4); block tile 128x256; wave tile 64x64; K-step 32.
// A/B K-tiles double-buffered in LDS via async copies; WMMA feeds from LDS.
// LDS rows padded 32->40 bf16 (80B) to break bank alignment.
template <int KDIM, int NDIM>
__global__ void __launch_bounds__(256)
gemm_bf16_kernel(const __bf16* __restrict__ A, const __bf16* __restrict__ W,
                 float* __restrict__ C, int M) {
    __shared__ __bf16 As[2][128][40];    // 10,240B per buffer
    __shared__ __bf16 Bs[2][256][40];    // 20,480B per buffer (61,440B total)

    const int wave = threadIdx.x >> 5;
    const int lane = threadIdx.x & 31;
    const int lhl = lane & 15, lhh = lane >> 4;
    const int mblk = blockIdx.y * 128;
    const int nblk = blockIdx.x * 256;
    const int m0w = (wave >> 2) * 64;    // wave tile inside block: 64(M) x 64(N)
    const int n0w = (wave & 3) * 64;

    // cooperative async tile load: A 512 b128-chunks, B 1024 b128-chunks
    auto load_tile = [&](int kt, int buf) {
        int ca = (int)threadIdx.x * 2;
#pragma unroll
        for (int i = 0; i < 2; ++i, ++ca) {
            int row = ca >> 2, j = ca & 3;
            async_copy_b128(&As[buf][row][j * 8],
                            A + (size_t)(mblk + row) * KDIM + kt * 32 + j * 8);
        }
        int cb = (int)threadIdx.x * 4;
#pragma unroll
        for (int i = 0; i < 4; ++i, ++cb) {
            int row = cb >> 2, j = cb & 3;
            async_copy_b128(&Bs[buf][row][j * 8],
                            W + (size_t)(nblk + row) * KDIM + kt * 32 + j * 8);
        }
    };

    v8f acc[4][4];
#pragma unroll
    for (int i = 0; i < 4; ++i)
#pragma unroll
        for (int j = 0; j < 4; ++j)
#pragma unroll
            for (int r = 0; r < 8; ++r) acc[i][j][r] = 0.0f;

    constexpr int NK = KDIM / 32;
    load_tile(0, 0);
    wait_async0();
    __syncthreads();

    for (int kt = 0; kt < NK; ++kt) {
        const int buf = kt & 1;
        if (kt + 1 < NK) load_tile(kt + 1, buf ^ 1);   // prefetch next K tile

        // fragments from LDS: 4 A rows x 4 B cols
        v16bf a[4], bw[4];
#pragma unroll
        for (int i = 0; i < 4; ++i) {
            const __bf16* p = &As[buf][m0w + i * 16 + lhl][lhh * 8];
#pragma unroll
            for (int e = 0; e < 8; ++e) a[i][e] = p[e];
#pragma unroll
            for (int e = 0; e < 8; ++e) a[i][8 + e] = p[16 + e];
        }
#pragma unroll
        for (int j = 0; j < 4; ++j) {
            const __bf16* p = &Bs[buf][n0w + j * 16 + lhl][lhh * 8];
#pragma unroll
            for (int e = 0; e < 16; ++e) bw[j][e] = p[e + (e >> 3) * 8];
        }
#pragma unroll
        for (int i = 0; i < 4; ++i)
#pragma unroll
            for (int j = 0; j < 4; ++j)
                acc[i][j] = __builtin_amdgcn_wmma_f32_16x16x32_bf16(
                    false, a[i], false, bw[j], (short)0, acc[i][j], false, false);

        wait_async0();       // next tile landed (no-op on last iteration)
        __syncthreads();     // everyone done reading buf; buf^1 complete
    }

#pragma unroll
    for (int i = 0; i < 4; ++i)
#pragma unroll
        for (int j = 0; j < 4; ++j) {
            float* cb = C + (size_t)(mblk + m0w + i * 16 + lhh * 8) * NDIM
                          + nblk + n0w + j * 16 + lhl;
#pragma unroll
            for (int r = 0; r < 8; ++r) cb[(size_t)r * NDIM] = acc[i][j][r];
        }
}

// ---------------------------------------------------------------------------
// RoPE + paged scatter. One thread per (b,s,head,pair). Q is pre-scaled by
// 1/sqrt(HD) so attention skips the score multiply. K/V scatter goes through
// block_table into the paged bf16 cache scratch (slot-major [slot][kv][hd]).
__global__ void __launch_bounds__(256)
rope_scatter_kernel(const float* __restrict__ qkv, const float* __restrict__ fc,
                    const int* __restrict__ btab,
                    __bf16* __restrict__ qb, __bf16* __restrict__ kp,
                    __bf16* __restrict__ vp) {
    size_t idx = (size_t)blockIdx.x * blockDim.x + threadIdx.x;
    const size_t total = (size_t)B_ * S_ * 48 * 64;
    if (idx >= total) return;
    const int d2 = idx & 63;
    size_t t = idx >> 6;                 // (b*S + s)*48 + h
    const int h = (int)(t % 48);
    const size_t bs = t / 48;            // b*S + s
    const int s = (int)(bs & (S_ - 1));
    const int b = (int)(bs >> 11);

    const float cv = fc[((size_t)s * 64 + d2) * 2 + 0];
    const float sv = fc[((size_t)s * 64 + d2) * 2 + 1];
    const float* src = qkv + bs * QKVC_ + h * HD_ + 2 * d2;
    float x0 = src[0], x1 = src[1];
    float r0 = x0, r1 = x1;
    if (h < NH_ + NKV_) {                // rotate q and k heads
        r0 = x0 * cv - x1 * sv;
        r1 = x1 * cv + x0 * sv;
    }
    if (h < NH_) {                       // Q: [b][h][s][hd], pre-scaled
        __bf16* dst = qb + (((size_t)(b * NH_ + h)) * S_ + s) * HD_ + 2 * d2;
        dst[0] = f2bf(r0 * SCALE_);
        dst[1] = f2bf(r1 * SCALE_);
    } else {
        const int blk = btab[b * NBLK_ + (s >> 4)];
        const int slot = blk * 16 + (s & 15);
        if (h < NH_ + NKV_) {            // K page
            __bf16* dst = kp + ((size_t)slot * NKV_ + (h - NH_)) * HD_ + 2 * d2;
            dst[0] = f2bf(r0);
            dst[1] = f2bf(r1);
        } else {                         // V page
            __bf16* dst = vp + ((size_t)slot * NKV_ + (h - NH_ - NKV_)) * HD_ + 2 * d2;
            dst[0] = f2bf(r0);
            dst[1] = f2bf(r1);
        }
    }
}

// ---------------------------------------------------------------------------
// Flash attention: one wave per 16-row Q tile per (b, head). Online softmax
// over 32-key chunks. K fragments load straight from the paged cache
// (two 16B runs/lane). V tiles are async-copied into a per-wave LDS tile
// (coalesced b128 rows, overlapped with QK^T + softmax), then B-fragments
// read from LDS. P round-trips C-layout -> A-layout through 1KB of LDS.
__global__ void __launch_bounds__(128)
attn_kernel(const __bf16* __restrict__ qb, const __bf16* __restrict__ kp,
            const __bf16* __restrict__ vp, const int* __restrict__ btab,
            const int* __restrict__ seqlens, __bf16* __restrict__ attn_out) {
    __shared__ __bf16 Vs[4][32][136];    // padded 272B row stride, 8.5KB/wave
    __shared__ __bf16 plds[4][16 * 32];
    const int wave = threadIdx.x >> 5;
    const int lane = threadIdx.x & 31;
    const int lhl = lane & 15, lhh = lane >> 4;

    const int tile = blockIdx.x * 4 + wave;        // B*NH*(S/16) tiles
    const int qt = tile & 127;
    const int h = (tile >> 7) & (NH_ - 1);
    const int b = tile >> 12;
    const int kvh = h >> 2;                        // GQA group
    const int seqlen = seqlens[b];

    // Q fragments for all 4 HD chunks (K=32 each)
    const __bf16* qbase = qb + ((size_t)(b * NH_ + h)) * S_ * HD_;
    v16bf aq[4];
#pragma unroll
    for (int kk = 0; kk < 4; ++kk)
        aq[kk] = load_a_frag(qbase, HD_, qt * 16, kk * 32, lane);

    float m[8], lsum[8];
    v8f acc[8];
#pragma unroll
    for (int r = 0; r < 8; ++r) { m[r] = -1e30f; lsum[r] = 0.0f; }
#pragma unroll
    for (int t = 0; t < 8; ++t)
#pragma unroll
        for (int r = 0; r < 8; ++r) acc[t][r] = 0.0f;

    const int qhi = qt * 16 + 15;
    for (int ks = 0; ks <= qhi; ks += 32) {
        // ---- stage V tile: lane -> one key row, 16 async b128 copies
        {
            const int blk = btab[b * NBLK_ + (ks >> 4) + lhh];
            const int slot = blk * 16 + lhl;       // key = ks + lane
            const __bf16* vrow = vp + ((size_t)slot * NKV_ + kvh) * HD_;
            __bf16* lrow = &Vs[wave][lane][0];
#pragma unroll
            for (int j = 0; j < 16; ++j)
                async_copy_b128(lrow + j * 8, vrow + j * 8);
        }

        // ---- S = Q K^T : two 16-key subtiles, 4 HD chunks -> 8 WMMA
        v8f c0, c1;
#pragma unroll
        for (int r = 0; r < 8; ++r) { c0[r] = 0.0f; c1[r] = 0.0f; }
        v16bf bk[2][4];
#pragma unroll
        for (int n = 0; n < 2; ++n) {
            const int n0 = ks + n * 16;
            const int blk = btab[b * NBLK_ + (n0 >> 4)];
            const __bf16* kb = kp + (((size_t)(blk * 16 + lhl)) * NKV_ + kvh) * HD_;
#pragma unroll
            for (int kk = 0; kk < 4; ++kk) {
                const __bf16* p = kb + kk * 32 + lhh * 8;
#pragma unroll
                for (int e = 0; e < 16; ++e) bk[n][kk][e] = p[e + (e >> 3) * 8];
            }
        }
#pragma unroll
        for (int kk = 0; kk < 4; ++kk) {
            c0 = __builtin_amdgcn_wmma_f32_16x16x32_bf16(
                false, aq[kk], false, bk[0][kk], (short)0, c0, false, false);
            c1 = __builtin_amdgcn_wmma_f32_16x16x32_bf16(
                false, aq[kk], false, bk[1][kk], (short)0, c1, false, false);
        }

        // ---- mask + online softmax (Q already pre-scaled by 1/sqrt(HD))
#pragma unroll
        for (int r = 0; r < 8; ++r) {
            const int q = qt * 16 + r + lhh * 8;
            const int k0 = ks + lhl, k1 = ks + 16 + lhl;
            float s0 = (k0 <= q && k0 < seqlen) ? c0[r] : -1e30f;
            float s1 = (k1 <= q && k1 < seqlen) ? c1[r] : -1e30f;
            float mnew = fmaxf(m[r], rmax16(fmaxf(s0, s1)));
            float corr = __expf(m[r] - mnew);
            float p0 = __expf(s0 - mnew);
            float p1 = __expf(s1 - mnew);
            lsum[r] = lsum[r] * corr + rsum16(p0 + p1);
            m[r] = mnew;
#pragma unroll
            for (int t = 0; t < 8; ++t) acc[t][r] *= corr;
            const int row = r + lhh * 8;
            plds[wave][row * 32 + lhl]      = f2bf(p0);
            plds[wave][row * 32 + 16 + lhl] = f2bf(p1);
        }

        // ---- reload P as A-fragment (same wave, DS ops are in-order)
        v16bf pa;
        {
            const __bf16* p = &plds[wave][lhl * 32 + lhh * 8];
#pragma unroll
            for (int e = 0; e < 8; ++e) pa[e] = p[e];
#pragma unroll
            for (int e = 0; e < 8; ++e) pa[8 + e] = p[16 + e];
        }

        // ---- O += P V : V B-fragments from the staged LDS tile, 8 WMMA
        wait_async0();
#pragma unroll
        for (int t = 0; t < 8; ++t) {
            v16bf bv;
            const __bf16* base = &Vs[wave][lhh * 8][t * 16 + lhl];
#pragma unroll
            for (int e = 0; e < 16; ++e) bv[e] = base[(size_t)(e + (e >> 3) * 8) * 136];
            acc[t] = __builtin_amdgcn_wmma_f32_16x16x32_bf16(
                false, pa, false, bv, (short)0, acc[t], false, false);
        }
    }

    // ---- normalize and emit bf16 rows [b*S + q][h*HD + hd] for the WO GEMM
#pragma unroll
    for (int r = 0; r < 8; ++r) {
        const float inv = 1.0f / lsum[r];
        const int q = qt * 16 + r + lhh * 8;
        const size_t row = (size_t)b * S_ + q;
#pragma unroll
        for (int t = 0; t < 8; ++t)
            attn_out[row * DIM_ + h * HD_ + t * 16 + lhl] = f2bf(acc[t][r] * inv);
    }
}

// ---------------------------------------------------------------------------
extern "C" void kernel_launch(void* const* d_in, const int* in_sizes, int n_in,
                              void* d_out, int out_size, void* d_ws, size_t ws_size,
                              hipStream_t stream) {
    const float* x    = (const float*)d_in[0];
    const float* wqkv = (const float*)d_in[1];
    const float* wo   = (const float*)d_in[2];
    const float* fc   = (const float*)d_in[3];
    // d_in[4], d_in[5]: zero-initialized caches (functional in the reference;
    // we materialize the paged cache in scratch instead of mutating inputs)
    const int* btab = (const int*)d_in[6];
    const int* seql = (const int*)d_in[7];
    float* out = (float*)d_out;

    char* ws = (char*)d_ws;
    __bf16* xb    = (__bf16*)(ws + 0);             //  33,554,432 B
    __bf16* wqkvb = (__bf16*)(ws + 33554432ull);   //  50,331,648 B
    __bf16* wob   = (__bf16*)(ws + 83886080ull);   //  33,554,432 B
    float*  qkv   = (float*) (ws + 117440512ull);  // 100,663,296 B
    __bf16* qbuf  = (__bf16*)(ws + 218103808ull);  //  33,554,432 B
    __bf16* kpage = (__bf16*)(ws + 251658240ull);  //   8,388,608 B
    __bf16* vpage = (__bf16*)(ws + 260046848ull);  //   8,388,608 B
    __bf16* attnb = (__bf16*)(ws + 268435456ull);  //  33,554,432 B (tot ~302MB)

    // 1) fp32 -> bf16 casts
    cast_f32_bf16_kernel<<<4096, 256, 0, stream>>>(x, xb, (size_t)B_ * S_ * DIM_);
    cast_f32_bf16_kernel<<<4096, 256, 0, stream>>>(wqkv, wqkvb, (size_t)QKVC_ * DIM_);
    cast_f32_bf16_kernel<<<4096, 256, 0, stream>>>(wo, wob, (size_t)DIM_ * DIM_);

    // 2) fused QKV projection: [B*S,DIM] @ [QKVC,DIM]^T -> fp32 scratch
    dim3 g1(QKVC_ / 256, (B_ * S_) / 128);
    gemm_bf16_kernel<DIM_, QKVC_><<<g1, 256, 0, stream>>>(xb, wqkvb, qkv, B_ * S_);

    // 3) RoPE + paged K/V scatter + Q pre-scale
    rope_scatter_kernel<<<49152, 256, 0, stream>>>(qkv, fc, btab, qbuf, kpage, vpage);

    // 4) GQA causal flash attention (8192 wave-tiles, 4 waves/block)
    attn_kernel<<<2048, 128, 0, stream>>>(qbuf, kpage, vpage, btab, seql, attnb);

    // 5) output projection: [B*S,DIM] @ [DIM,DIM]^T -> d_out (fp32)
    dim3 g2(DIM_ / 256, (B_ * S_) / 128);
    gemm_bf16_kernel<DIM_, DIM_><<<g2, 256, 0, stream>>>(attnb, wob, out, B_ * S_);
}
